// AttentionModuleEx1_25683904430243
// MI455X (gfx1250) — matrix operand order repaired and checked by reading, not compiled, not run on hardware
//
#include <hip/hip_runtime.h>

// Problem constants (from reference): x is (B=8, C=512, H=64, W=64) fp32.
#define CDIM 512
#define HH   64
#define WW   64
#define HW   4096            // HH*WW
#define NBAT 8
#define NE   (NBAT * CDIM * HW)   // 16,777,216 elements

typedef __bf16        v16bf __attribute__((ext_vector_type(16)));
typedef float         v8f   __attribute__((ext_vector_type(8)));
typedef unsigned int  v8u   __attribute__((ext_vector_type(8)));

__device__ __forceinline__ unsigned short f2bf(float f) {
    // round-to-nearest-even fp32 -> bf16
    unsigned int u = __float_as_uint(f);
    u += 0x7fffu + ((u >> 16) & 1u);
    return (unsigned short)(u >> 16);
}

// ---------------------------------------------------------------------------
// Kernel 1: fully fused depthwise chain, one (n,c) 64x64 plane per block.
// All seven depthwise convs are zero-padded and stay inside the plane, so the
// whole chain runs out of LDS: read x once, write bf16(attn-sum) once.
// LDS: 4 x 16KB buffers = 64KB (WGP has 320KB -> ~5 blocks resident).
// ---------------------------------------------------------------------------

template <int K>
__device__ __forceinline__ float vconv_px(const float* __restrict__ in,
                                          const float* __restrict__ w,
                                          float bb, int y, int xx) {
    const int pad = 3 * (K - 1) / 2;
    float acc = bb;
#pragma unroll
    for (int j = 0; j < K; ++j) {
        int ys = y + 3 * j - pad;
        if (ys >= 0 && ys < HH) acc = fmaf(w[j], in[ys * WW + xx], acc);
    }
    return acc;
}

template <int K>
__device__ __forceinline__ void hconv(const float* __restrict__ in,
                                      float* __restrict__ outb,
                                      const float* __restrict__ w,
                                      float bb, int tid) {
    const int pad = 3 * (K - 1) / 2;
#pragma unroll
    for (int k = 0; k < HW / 256; ++k) {
        int i = k * 256 + tid, y = i >> 6, xx = i & 63;
        float acc = bb;
#pragma unroll
        for (int j = 0; j < K; ++j) {
            int xs = xx + 3 * j - pad;
            if (xs >= 0 && xs < WW) acc = fmaf(w[j], in[y * WW + xs], acc);
        }
        outb[i] = acc;
    }
}

__global__ __launch_bounds__(256)
void lka_dw_fused(const float* __restrict__ x,
                  const float* __restrict__ w0,  const float* __restrict__ b0,
                  const float* __restrict__ w01, const float* __restrict__ b01,
                  const float* __restrict__ w02, const float* __restrict__ b02,
                  const float* __restrict__ w11, const float* __restrict__ b11,
                  const float* __restrict__ w12, const float* __restrict__ b12,
                  const float* __restrict__ w21, const float* __restrict__ b21,
                  const float* __restrict__ w22, const float* __restrict__ b22,
                  unsigned short* __restrict__ actb) {
    __shared__ float buf0[HW];   // x plane, later reused for h21 output
    __shared__ float buf1[HW];   // attn (5x5 output)
    __shared__ float buf2[HW];   // h7 output
    __shared__ float buf3[HW];   // h11 output

    const int c   = blockIdx.x;
    const int nb  = blockIdx.y;
    const int tid = threadIdx.x;
    const size_t gbase = ((size_t)nb * CDIM + c) * HW;

    // ---- load channel plane (coalesced, 16 px/thread) ----
#pragma unroll
    for (int k = 0; k < HW / 256; ++k)
        buf0[k * 256 + tid] = x[gbase + k * 256 + tid];
    __syncthreads();

    // ---- 5x5 depthwise, pad 2 ----
    {
        const float* wc = w0 + c * 25;
        const float  bb = b0[c];
#pragma unroll
        for (int k = 0; k < HW / 256; ++k) {
            int i = k * 256 + tid, y = i >> 6, xx = i & 63;
            float acc = bb;
#pragma unroll
            for (int dy = -2; dy <= 2; ++dy) {
                int ys = y + dy;
                if (ys < 0 || ys >= HH) continue;
#pragma unroll
                for (int dx = -2; dx <= 2; ++dx) {
                    int xs = xx + dx;
                    if (xs >= 0 && xs < WW)
                        acc = fmaf(wc[(dy + 2) * 5 + (dx + 2)], buf0[ys * WW + xs], acc);
                }
            }
            buf1[i] = acc;
        }
    }
    __syncthreads();   // buf1 ready; all buf0 reads done -> buf0 reusable

    // ---- horizontal dilated(3) convs of attn ----
    hconv<7 >(buf1, buf2, w01 + c * 7,  b01[c], tid);
    hconv<11>(buf1, buf3, w11 + c * 11, b11[c], tid);
    hconv<21>(buf1, buf0, w21 + c * 21, b21[c], tid);
    __syncthreads();

    // ---- vertical dilated(3) convs + branch sum, emit bf16 ----
    {
        const float* wv7  = w02 + c * 7;   const float bv7  = b02[c];
        const float* wv11 = w12 + c * 11;  const float bv11 = b12[c];
        const float* wv21 = w22 + c * 21;  const float bv21 = b22[c];
#pragma unroll
        for (int k = 0; k < HW / 256; ++k) {
            int i = k * 256 + tid, y = i >> 6, xx = i & 63;
            float s = buf1[i]
                    + vconv_px<7 >(buf2, wv7,  bv7,  y, xx)
                    + vconv_px<11>(buf3, wv11, bv11, y, xx)
                    + vconv_px<21>(buf0, wv21, bv21, y, xx);
            actb[gbase + i] = f2bf(s);
        }
    }
}

// ---------------------------------------------------------------------------
// Kernel 2: convert the 512x512 1x1-conv weight to bf16 (0.5 MB, trivial).
// ---------------------------------------------------------------------------
__global__ __launch_bounds__(256)
void cvt_w3_bf16(const float* __restrict__ w3, unsigned short* __restrict__ w3b) {
    int i = blockIdx.x * 256 + threadIdx.x;
    w3b[i] = f2bf(w3[i]);
}

// ---------------------------------------------------------------------------
// Kernel 3: 1x1 conv as bf16 WMMA GEMM with fused bias + gating epilogue.
//   out[nb, co, p] = (sum_ci w3[co,ci] * act[nb,ci,p] + b3[co]) * x[nb,co,p]
// Per wave: one 16(co) x 16(p) tile, K=512 in 16 steps of
// v_wmma_f32_16x16x32_bf16 (unrolled x4). Fragment packing per CDNA5 ISA:
//   A (16x32, M x K): lane m=L&15, half=L>>4; VGPR v holds K = 16*(v>>2)+8*half+2*(v&3), +1
//     -> VGPRs 0..3 and 4..7 are each 16 contiguous bytes of a weight row:
//        one uint4 (b128) load each.
//   B (32x16, K x N): lane n=L&15, half=L>>4; VGPR v holds K = 16*(v>>2)+8*half+2*(v&3), +1
//     -> ushort2 per VGPR (d16 / d16_hi loads, K pairs HW apart in memory).
//   C/D (16x16 f32):  lane n=L&15; VGPR r holds M = r + 8*(L>>4)
// ---------------------------------------------------------------------------
__global__ __launch_bounds__(256)
void gemm1x1_bias_mul(const unsigned short* __restrict__ actb,  // [NB][C][HW] bf16
                      const unsigned short* __restrict__ w3b,   // [C][C] bf16
                      const float* __restrict__ b3,
                      const float* __restrict__ xin,
                      float* __restrict__ out) {
    const int lane = threadIdx.x & 31;
    const int wv   = threadIdx.x >> 5;       // wave id in block: 0..7
    const int half = lane >> 4;
    const int l16  = lane & 15;

    const int p0  = (blockIdx.x * 8 + wv) * 16;   // pixel-tile origin (0..4080)
    const int co0 = blockIdx.y * 16;              // out-channel tile origin
    const int nb  = blockIdx.z;

    const unsigned short* act = actb + (size_t)nb * CDIM * HW;

    v8f acc = {};
    // weight row for this lane; 16-byte aligned at every K step used below
    const unsigned short* wrow = w3b + (size_t)(co0 + l16) * CDIM + 8 * half;

    union AFrag { v8u v; uint4 q[2]; };
    union BFrag { v8u v; ushort2 s[8]; };

#pragma unroll 4
    for (int kb = 0; kb < CDIM; kb += 32) {
        // A operand: two b128 loads cover K = kb+8*half+0..7 and +16..23
        AFrag a;
        a.q[0] = *(const uint4*)(wrow + kb);        // VGPRs 0..3
        a.q[1] = *(const uint4*)(wrow + kb + 16);   // VGPRs 4..7
        // B operand: K pairs are HW apart in memory (d16/d16_hi pattern)
        BFrag b;
#pragma unroll
        for (int v = 0; v < 8; ++v) {
            int ci = kb + 16 * (v >> 2) + 8 * half + 2 * (v & 3);
            b.s[v].x = act[(size_t)ci * HW + p0 + l16];
            b.s[v].y = act[(size_t)(ci + 1) * HW + p0 + l16];
        }
        acc = __builtin_amdgcn_wmma_f32_16x16x32_bf16(
                  /*neg_a=*/false, __builtin_bit_cast(v16bf, a.v),
                  /*neg_b=*/false, __builtin_bit_cast(v16bf, b.v),
                  /*c_mod=*/(short)0, acc,
                  /*reuse_a=*/false, /*reuse_b=*/false);
    }

    // Epilogue: bias + elementwise gate by original x.
    const size_t obase = (size_t)nb * CDIM * HW;
#pragma unroll
    for (int r = 0; r < 8; ++r) {
        int co = co0 + r + 8 * half;
        size_t idx = obase + (size_t)co * HW + p0 + l16;
        float v = acc[r] + b3[co];
        out[idx] = v * xin[idx];
    }
}

// ---------------------------------------------------------------------------
// Launch. Inputs (setup_inputs order):
//  0:x 1:w0 2:b0 3:w0_1 4:b0_1 5:w0_2 6:b0_2 7:w1_1 8:b1_1 9:w1_2 10:b1_2
//  11:w2_1 12:b2_1 13:w2_2 14:b2_2 15:w3 16:b3
// Workspace: NE bf16 activations + 512*512 bf16 weights = 34,078,720 bytes.
// ---------------------------------------------------------------------------
extern "C" void kernel_launch(void* const* d_in, const int* in_sizes, int n_in,
                              void* d_out, int out_size, void* d_ws, size_t ws_size,
                              hipStream_t stream) {
    const float* x   = (const float*)d_in[0];
    const float* w0  = (const float*)d_in[1];  const float* b0  = (const float*)d_in[2];
    const float* w01 = (const float*)d_in[3];  const float* b01 = (const float*)d_in[4];
    const float* w02 = (const float*)d_in[5];  const float* b02 = (const float*)d_in[6];
    const float* w11 = (const float*)d_in[7];  const float* b11 = (const float*)d_in[8];
    const float* w12 = (const float*)d_in[9];  const float* b12 = (const float*)d_in[10];
    const float* w21 = (const float*)d_in[11]; const float* b21 = (const float*)d_in[12];
    const float* w22 = (const float*)d_in[13]; const float* b22 = (const float*)d_in[14];
    const float* w3  = (const float*)d_in[15]; const float* b3  = (const float*)d_in[16];
    float* out = (float*)d_out;

    unsigned short* actb = (unsigned short*)d_ws;     // NE bf16
    unsigned short* w3b  = actb + (size_t)NE;         // 512*512 bf16

    // 1) fused depthwise chain: one block per (c, n) plane
    lka_dw_fused<<<dim3(CDIM, NBAT), 256, 0, stream>>>(
        x, w0, b0, w01, b01, w02, b02, w11, b11, w12, b12, w21, b21, w22, b22, actb);

    // 2) weight conversion for the WMMA GEMM
    cvt_w3_bf16<<<(CDIM * CDIM) / 256, 256, 0, stream>>>(w3, w3b);

    // 3) 1x1 channel mix + bias + gate: 8 waves/block, 16x16 tile per wave
    gemm1x1_bias_mul<<<dim3(HW / 16 / 8, CDIM / 16, NBAT), 256, 0, stream>>>(
        actb, w3b, b3, x, out);
}
